// MultiHeadAttention_17583596110389
// MI455X (gfx1250) — matrix operand order repaired and checked by reading, not compiled, not run on hardware
//
#include <hip/hip_runtime.h>
#include <hip/hip_bf16.h>

// Problem constants
#define BDIM 8
#define TDIM 1024
#define EDIM 256
#define HDIM 8
#define NE   (HDIM * EDIM)   // 2048
#define MROWS (BDIM * TDIM)  // 8192
#define QTILE 64
#define KTILE 32

// Padded LDS strides (elements) to reduce 64-bank conflicts; all keep 16B row alignment.
#define KSTRIDE 40    // K^T tile rows ([dim][key]), 40*2B = 80B
#define VSTRIDE 264   // V tile rows ([key][dim]), 264*2B = 528B
#define PSTRIDE 40    // P staging rows
#define ASTRIDE 40    // GEMM A tiles
#define BSTR_Q  136   // qkv GEMM B tile (32x128)
#define BSTR_F  72    // fc GEMM B tile (32x64)

typedef __bf16 bf16x16 __attribute__((ext_vector_type(16)));
typedef float  f32x8   __attribute__((ext_vector_type(8)));

union FragAB {
    unsigned int u[8];
    bf16x16 b;
};

#if defined(__HIP_DEVICE_COMPILE__) && __has_builtin(__builtin_amdgcn_global_load_async_to_lds_b128)
#define HAVE_ASYNC_LDS 1
typedef int v4i __attribute__((ext_vector_type(4)));
typedef __attribute__((address_space(1))) v4i glb_v4i;   // HIP prints AS1 as __device__
typedef __attribute__((address_space(3))) v4i lds_v4i;   // HIP prints AS3 as __shared__
// Route casts through integers: flat->LDS is addr[31:0] per ISA aperture rules.
__device__ __forceinline__ void async_copy16(const void* g, void* s) {
    __builtin_amdgcn_global_load_async_to_lds_b128(
        (glb_v4i*)(unsigned long long)g,
        (lds_v4i*)(unsigned int)(unsigned long long)s,
        0, 0);
}
#define ASYNC_COPY16(gp, sp) async_copy16((const void*)(gp), (void*)(sp))
#define ASYNC_WAIT() asm volatile("s_wait_asynccnt 0x0" ::: "memory")
#else
#define HAVE_ASYNC_LDS 0
#define ASYNC_COPY16(gp, sp) (*(sp) = *(gp))
#define ASYNC_WAIT()
#endif

__device__ __forceinline__ unsigned short f2bf(float x) {
    unsigned int v = __float_as_uint(x);
    v += 0x7FFFu + ((v >> 16) & 1u);      // round-to-nearest-even
    return (unsigned short)(v >> 16);
}

// A fragment: 16(M) x 32(K) bf16 from row-major buffer (stride in elements).
// Lane l: row = l&15, K-group g = l>>4. v0..3 -> K = 2v + 8g ; v4..7 -> K = 16 + 2(v-4) + 8g.
__device__ __forceinline__ FragAB load_frag_a(const unsigned short* base, int stride) {
    int lane = threadIdx.x & 31;
    int r = lane & 15, g = lane >> 4;
    const unsigned short* p = base + r * stride;
    FragAB f;
#pragma unroll
    for (int v = 0; v < 8; ++v) {
        int kk = (v < 4) ? (2 * v + 8 * g) : (16 + 2 * (v - 4) + 8 * g);
        f.u[v] = *(const unsigned int*)(p + kk);
    }
    return f;
}

// B fragment: 32(K) x 16(N) bf16 from row-major KxN buffer.
// Lane: col = l&15, K-group g = l>>4. v0..3 -> K = 2v + 8g ; v4..7 -> K = 16 + 2(v-4) + 8g.
__device__ __forceinline__ FragAB load_frag_b(const unsigned short* base, int stride) {
    int lane = threadIdx.x & 31;
    int n = lane & 15, g = lane >> 4;
    const unsigned short* p = base + n;
    FragAB f;
#pragma unroll
    for (int v = 0; v < 8; ++v) {
        int kk = (v < 4) ? (2 * v + 8 * g) : (16 + 2 * (v - 4) + 8 * g);
        f.u[v] = (unsigned int)p[kk * stride] | ((unsigned int)p[(kk + 1) * stride] << 16);
    }
    return f;
}

__device__ __forceinline__ f32x8 wmma_bf16(FragAB a, FragAB b, f32x8 c) {
    return __builtin_amdgcn_wmma_f32_16x16x32_bf16(false, a.b, false, b.b,
                                                   (short)0, c, false, false);
}

// ---------------------------------------------------------------------------
// Kernel 1: QKV projection GEMM.  C[M=8192,N=2048] = X[8192,256] @ W[256,2048]
// kt=0: out bf16 [B,H,T,E]  (Q scaled by 1/sqrt(E), V)
// kt=1: out bf16 [B,H,E,T]  (K stored transposed for the attention kernel)
// Block: 128x128 tile, 256 threads = 8 waves (4 along M x 2 along N), wave 32x64.
// ---------------------------------------------------------------------------
__global__ __launch_bounds__(256) void qkv_gemm_kernel(
    const float* __restrict__ X, const float* __restrict__ W,
    unsigned short* __restrict__ out, float scale, int kt)
{
    __shared__ __align__(16) unsigned short As[128 * ASTRIDE];
    __shared__ __align__(16) unsigned short Bs[32 * BSTR_Q];
    const int m0 = blockIdx.y * 128;
    const int n0 = blockIdx.x * 128;
    const int tid = threadIdx.x;
    const int wave = tid >> 5;
    const int wm = (wave >> 1) * 32;
    const int wn = (wave & 1) * 64;

    f32x8 acc[2][4] = {};

    for (int k0 = 0; k0 < EDIM; k0 += 32) {
        __syncthreads();
#pragma unroll
        for (int i = 0; i < 16; ++i) {   // A tile 128x32, f32 -> bf16
            int idx = tid + i * 256;
            int r = idx >> 5, c = idx & 31;
            As[r * ASTRIDE + c] = f2bf(X[(size_t)(m0 + r) * EDIM + k0 + c]);
        }
#pragma unroll
        for (int i = 0; i < 16; ++i) {   // B tile 32x128, f32 -> bf16
            int idx = tid + i * 256;
            int r = idx >> 7, c = idx & 127;
            Bs[r * BSTR_Q + c] = f2bf(W[(size_t)(k0 + r) * NE + n0 + c]);
        }
        __syncthreads();

        FragAB a0 = load_frag_a(As + (wm + 0) * ASTRIDE, ASTRIDE);
        FragAB a1 = load_frag_a(As + (wm + 16) * ASTRIDE, ASTRIDE);
        FragAB bfr[4];
#pragma unroll
        for (int j = 0; j < 4; ++j) bfr[j] = load_frag_b(Bs + wn + 16 * j, BSTR_Q);
#pragma unroll
        for (int j = 0; j < 4; ++j) {
            acc[0][j] = wmma_bf16(a0, bfr[j], acc[0][j]);
            acc[1][j] = wmma_bf16(a1, bfr[j], acc[1][j]);
        }
    }

    const int lane = tid & 31;
    const int cn = lane & 15;
    const int rg = (lane >> 4) * 8;
#pragma unroll
    for (int i = 0; i < 2; ++i)
#pragma unroll
        for (int j = 0; j < 4; ++j)
#pragma unroll
            for (int v = 0; v < 8; ++v) {
                int m = m0 + wm + 16 * i + rg + v;
                int n = n0 + wn + 16 * j + cn;
                int b = m >> 10, t = m & (TDIM - 1);
                int h = n >> 8, e = n & (EDIM - 1);
                size_t o = kt ? (((size_t)(b * HDIM + h) * EDIM + e) * TDIM + t)
                              : (((size_t)(b * HDIM + h) * TDIM + t) * EDIM + e);
                out[o] = f2bf(acc[i][j][v] * scale);
            }
}

// ---------------------------------------------------------------------------
// Kernel 2: fused flash attention. One block = 64 q-rows of one (b,h).
// 4 waves x 16 q-rows; key-blocks of 32 double-buffered through LDS
// (async global->LDS when available). K arrives pre-transposed [B,H,E,T] so
// both WMMA B-operands are row-major [k][n] LDS tiles read by load_frag_b.
// ---------------------------------------------------------------------------
__global__ __launch_bounds__(128) void attn_kernel(
    const unsigned short* __restrict__ Qb,   // [B,H,T,E] bf16 (pre-scaled 1/16)
    const unsigned short* __restrict__ KbT,  // [B,H,E,T] bf16
    const unsigned short* __restrict__ Vb,   // [B,H,T,E] bf16
    const unsigned char* __restrict__ mask,  // [B,T,T] bool
    unsigned short* __restrict__ Ob)         // [B,T,H*E] bf16
{
    __shared__ __align__(16) unsigned short KsT[2][EDIM * KSTRIDE];   // [dim][key]
    __shared__ __align__(16) unsigned short Vs [2][KTILE * VSTRIDE];  // [key][dim]
    __shared__ __align__(16) unsigned short Ps [4 * 16 * PSTRIDE];    // per-wave P staging

    const int h = blockIdx.y, b = blockIdx.z;
    const int tid = threadIdx.x;
    const int wave = tid >> 5;
    const int lane = tid & 31;
    const int q0 = blockIdx.x * QTILE + wave * 16;
    const int cn = lane & 15;
    const int rg = (lane >> 4) * 8;
    const float NEGINF = -__builtin_inff();

    const unsigned short* qh  = Qb  + (size_t)(b * HDIM + h) * TDIM * EDIM;
    const unsigned short* khT = KbT + (size_t)(b * HDIM + h) * EDIM * TDIM;
    const unsigned short* vh  = Vb  + (size_t)(b * HDIM + h) * TDIM * EDIM;
    const unsigned char* mp = mask + (size_t)b * TDIM * TDIM;

    // Copy one key-block (K^T 256x32, V 32x128) into LDS buffer `buf`.
    auto copy_tiles = [&](int buf, int kb) {
        const unsigned short* kg0 = khT + kb;
#pragma unroll
        for (int i = 0; i < 8; ++i) {            // 256 dims x 4 x 16B chunks
            int c = tid + i * 128;
            int row = c >> 2, sub = c & 3;
            const ulonglong2* g = (const ulonglong2*)(kg0 + (size_t)row * TDIM);
            ulonglong2* s = (ulonglong2*)(&KsT[buf][row * KSTRIDE]);
            ASYNC_COPY16(g + sub, s + sub);
        }
        const unsigned short* vg0 = vh + (size_t)kb * EDIM;
#pragma unroll
        for (int i = 0; i < 4; ++i) {            // 32 keys x 16 x 16B chunks
            int c = tid + i * 128;
            int row = c >> 4, sub = c & 15;
            const ulonglong2* g = (const ulonglong2*)(vg0 + (size_t)row * EDIM);
            ulonglong2* s = (ulonglong2*)(&Vs[buf][row * VSTRIDE]);
            ASYNC_COPY16(g + sub, s + sub);
        }
    };

    // Resident Q fragments: 16 rows x 256 dims = 8 A-fragments.
    FragAB qf[8];
#pragma unroll
    for (int e = 0; e < 8; ++e)
        qf[e] = load_frag_a(qh + (size_t)q0 * EDIM + e * 32, EDIM);

    f32x8 oacc[16] = {};
    float mrow[8], lrow[8];
#pragma unroll
    for (int v = 0; v < 8; ++v) { mrow[v] = NEGINF; lrow[v] = 0.0f; }

    copy_tiles(0, 0);
    ASYNC_WAIT();
    __syncthreads();

    for (int kb = 0; kb < TDIM; kb += KTILE) {
        const int cur = (kb >> 5) & 1;
        if (kb + KTILE < TDIM)                    // prefetch next block into other buffer
            copy_tiles(cur ^ 1, kb + KTILE);

        // S = Q * K^T : 16 x 32 (two 16x16 C tiles), contraction over 256 dims
        f32x8 s0 = {}, s1 = {};
#pragma unroll
        for (int e = 0; e < 8; ++e) {
            FragAB b0 = load_frag_b(&KsT[cur][(e * 32) * KSTRIDE], KSTRIDE);      // keys 0..15
            FragAB b1 = load_frag_b(&KsT[cur][(e * 32) * KSTRIDE + 16], KSTRIDE); // keys 16..31
            s0 = wmma_bf16(qf[e], b0, s0);
            s1 = wmma_bf16(qf[e], b1, s1);
        }

        // mask + online softmax (C layout: lane = column, row = rg + v)
        float p0[8], p1[8], corr[8];
#pragma unroll
        for (int v = 0; v < 8; ++v) {
            int row = q0 + rg + v;
            float sv0 = s0[v], sv1 = s1[v];
            if (mp[(size_t)row * TDIM + kb + cn])      sv0 = NEGINF;
            if (mp[(size_t)row * TDIM + kb + 16 + cn]) sv1 = NEGINF;
            float mt = fmaxf(sv0, sv1);
            mt = fmaxf(mt, __shfl_xor(mt, 1));
            mt = fmaxf(mt, __shfl_xor(mt, 2));
            mt = fmaxf(mt, __shfl_xor(mt, 4));
            mt = fmaxf(mt, __shfl_xor(mt, 8));
            float mnew = fmaxf(mrow[v], mt);
            float cf = (mnew == NEGINF) ? 1.0f : __expf(mrow[v] - mnew);
            float e0 = (mnew == NEGINF) ? 0.0f : __expf(sv0 - mnew);
            float e1 = (mnew == NEGINF) ? 0.0f : __expf(sv1 - mnew);
            float ps = e0 + e1;
            ps += __shfl_xor(ps, 1);
            ps += __shfl_xor(ps, 2);
            ps += __shfl_xor(ps, 4);
            ps += __shfl_xor(ps, 8);
            lrow[v] = lrow[v] * cf + ps;
            mrow[v] = mnew;
            corr[v] = cf;
            p0[v] = e0; p1[v] = e1;
        }
#pragma unroll
        for (int nt = 0; nt < 16; ++nt)
#pragma unroll
            for (int v = 0; v < 8; ++v) oacc[nt][v] *= corr[v];

        // C-layout P -> bf16 -> per-wave LDS staging -> reload as A fragment
        unsigned short* psb = Ps + wave * 16 * PSTRIDE;
#pragma unroll
        for (int v = 0; v < 8; ++v) {
            int row = rg + v;
            psb[row * PSTRIDE + cn]      = f2bf(p0[v]);
            psb[row * PSTRIDE + 16 + cn] = f2bf(p1[v]);
        }
        asm volatile("s_wait_dscnt 0" ::: "memory");  // wave-internal DS RAW fence
        FragAB pf = load_frag_a(psb, PSTRIDE);

        // O += P * V  (16 output tiles covering 256 dims)
#pragma unroll
        for (int nt = 0; nt < 16; ++nt) {
            FragAB bv = load_frag_b(&Vs[cur][nt * 16], VSTRIDE);
            oacc[nt] = wmma_bf16(pf, bv, oacc[nt]);
        }

        ASYNC_WAIT();        // next-buffer copies complete before anyone reads them
        __syncthreads();
    }

    float inv[8];
#pragma unroll
    for (int v = 0; v < 8; ++v) inv[v] = (lrow[v] > 0.0f) ? 1.0f / lrow[v] : 0.0f;

    // store O as bf16 in [B, T, H*E] layout (rows of the final GEMM)
#pragma unroll
    for (int nt = 0; nt < 16; ++nt)
#pragma unroll
        for (int v = 0; v < 8; ++v) {
            int row = q0 + rg + v;
            int n = nt * 16 + cn;
            Ob[(size_t)(b * TDIM + row) * NE + h * EDIM + n] = f2bf(oacc[nt][v] * inv[v]);
        }
}

// ---------------------------------------------------------------------------
// Kernel 3: output projection.  out[8192,256] = O(bf16)[8192,2048] @ Wfc + bfc
// Block: 64x64 tile, 128 threads = 4 waves (2x2), wave 32x32.
// ---------------------------------------------------------------------------
__global__ __launch_bounds__(128) void fc_gemm_kernel(
    const unsigned short* __restrict__ A,
    const float* __restrict__ W,
    const float* __restrict__ bias,
    float* __restrict__ out)
{
    __shared__ __align__(16) unsigned short As[64 * ASTRIDE];
    __shared__ __align__(16) unsigned short Bs[32 * BSTR_F];
    const int m0 = blockIdx.y * 64;
    const int n0 = blockIdx.x * 64;
    const int tid = threadIdx.x;
    const int wave = tid >> 5;
    const int wm = (wave >> 1) * 32;
    const int wn = (wave & 1) * 32;

    f32x8 acc[2][2] = {};

    for (int k0 = 0; k0 < NE; k0 += 32) {
        __syncthreads();
#pragma unroll
        for (int i = 0; i < 8; ++i) {    // A tile 64x32 bf16 as u32 pairs
            int idx = tid + i * 128;
            int r = idx >> 4, c = idx & 15;
            ((unsigned int*)(As + r * ASTRIDE))[c] =
                ((const unsigned int*)(A + (size_t)(m0 + r) * NE + k0))[c];
        }
#pragma unroll
        for (int i = 0; i < 16; ++i) {   // B tile 32x64, f32 -> bf16
            int idx = tid + i * 128;
            int r = idx >> 6, c = idx & 63;
            Bs[r * BSTR_F + c] = f2bf(W[(size_t)(k0 + r) * EDIM + n0 + c]);
        }
        __syncthreads();

        FragAB a0 = load_frag_a(As + (wm + 0) * ASTRIDE, ASTRIDE);
        FragAB a1 = load_frag_a(As + (wm + 16) * ASTRIDE, ASTRIDE);
        FragAB b0 = load_frag_b(Bs + wn, BSTR_F);
        FragAB b1 = load_frag_b(Bs + wn + 16, BSTR_F);
        acc[0][0] = wmma_bf16(a0, b0, acc[0][0]);
        acc[0][1] = wmma_bf16(a0, b1, acc[0][1]);
        acc[1][0] = wmma_bf16(a1, b0, acc[1][0]);
        acc[1][1] = wmma_bf16(a1, b1, acc[1][1]);
    }

    const int lane = tid & 31;
    const int cn = lane & 15;
    const int rg = (lane >> 4) * 8;
#pragma unroll
    for (int i = 0; i < 2; ++i)
#pragma unroll
        for (int j = 0; j < 2; ++j) {
            int n = n0 + wn + 16 * j + cn;
            float bb = bias[n];
#pragma unroll
            for (int v = 0; v < 8; ++v) {
                int m = m0 + wm + 16 * i + rg + v;
                out[(size_t)m * EDIM + n] = acc[i][j][v] + bb;
            }
        }
}

// ---------------------------------------------------------------------------
extern "C" void kernel_launch(void* const* d_in, const int* in_sizes, int n_in,
                              void* d_out, int out_size, void* d_ws, size_t ws_size,
                              hipStream_t stream) {
    const float* x    = (const float*)d_in[0];
    const float* Wq   = (const float*)d_in[1];
    const float* Wk   = (const float*)d_in[2];
    const float* Wv   = (const float*)d_in[3];
    const float* Wfc  = (const float*)d_in[4];
    const float* bfc  = (const float*)d_in[5];
    const unsigned char* mask = (const unsigned char*)d_in[6];
    float* out = (float*)d_out;

    const size_t qbytes = (size_t)BDIM * HDIM * TDIM * EDIM * sizeof(unsigned short);
    unsigned short* qb = (unsigned short*)d_ws;
    unsigned short* kb = (unsigned short*)((char*)d_ws + 1 * qbytes);
    unsigned short* vb = (unsigned short*)((char*)d_ws + 2 * qbytes);
    unsigned short* ob = (unsigned short*)((char*)d_ws + 3 * qbytes);

    dim3 gq(NE / 128, MROWS / 128);            // (16, 64)
    qkv_gemm_kernel<<<gq, 256, 0, stream>>>(x, Wq, qb, 1.0f / 16.0f, 0); // Q * 1/sqrt(E)
    qkv_gemm_kernel<<<gq, 256, 0, stream>>>(x, Wk, kb, 1.0f, 1);          // K transposed
    qkv_gemm_kernel<<<gq, 256, 0, stream>>>(x, Wv, vb, 1.0f, 0);

    dim3 ga(TDIM / QTILE, HDIM, BDIM);         // (16, 8, 8)
    attn_kernel<<<ga, 128, 0, stream>>>(qb, kb, vb, mask, ob);

    dim3 gf(EDIM / 64, MROWS / 64);            // (4, 128)
    fc_gemm_kernel<<<gf, 128, 0, stream>>>(ob, Wfc, bfc, out);
}
